// StaticGraphEncoder_29540785062339
// MI455X (gfx1250) — matrix operand order repaired and checked by reading, not compiled, not run on hardware
//
#include <hip/hip_runtime.h>
#include <hip/hip_bf16.h>
#include <stdint.h>

#define NNODES 50000
#define NEDGES 400000
#define NPAD   50048      /* 391 * 128 */
#define NGRAPH 64
#define LEAKY  0.2f
#define BNEPS  1e-5f
#define NEGKEY 0x00800000u   /* key(-FLT_MAX): ordered-uint encoding */

typedef __attribute__((ext_vector_type(16))) _Float16 v16h;
typedef __attribute__((ext_vector_type(8)))  float    v8f;
typedef __attribute__((ext_vector_type(8)))  int      v8i;
typedef __attribute__((ext_vector_type(4)))  float    f32x4;

union ABFrag { v8i i; v16h h; };

__device__ __forceinline__ unsigned f2key(float f) {
  unsigned u = __float_as_uint(f);
  return (u & 0x80000000u) ? ~u : (u | 0x80000000u);
}
__device__ __forceinline__ float key2f(unsigned k) {
  unsigned u = (k & 0x80000000u) ? (k & 0x7fffffffu) : ~k;
  return __uint_as_float(u);
}

// ---------------- utility fills / converts ----------------
__global__ void k_fill_u32(uint32_t* p, uint32_t v, int n) {
  int i = blockIdx.x * blockDim.x + threadIdx.x;
  if (i < n) p[i] = v;
}

__global__ void k_cvt_f16(const float* __restrict__ x, _Float16* __restrict__ y, int n) {
  int i = blockIdx.x * blockDim.x + threadIdx.x;
  if (i < n) y[i] = (_Float16)x[i];
}

// ---------------- weight prep ----------------
// W: [3][128][128] (out,in) layer slice of lin_qkv; att: [3][128] layer slice (attq,attk,atte flat)
// wt: [3][128][128] f16, kin-major (transposed), q/k scaled by att vectors
__global__ void k_prep_qkv_w(const float* __restrict__ W, const float* __restrict__ att,
                             _Float16* __restrict__ wt) {
  int idx = blockIdx.x * blockDim.x + threadIdx.x;
  if (idx >= 3 * 128 * 128) return;
  int m   = idx >> 14;
  int rem = idx & 16383;
  int kin = rem >> 7;
  int j   = rem & 127;
  float s = (m == 2) ? 1.0f : att[m * 128 + j];
  wt[m * 16384 + kin * 128 + j] = (_Float16)(W[m * 16384 + j * 128 + kin] * s);
}

// We: [128][16] layer slice of lin_edge; atte: [128]; M: [16][8]
__global__ void k_prep_edge_mat(const float* __restrict__ We, const float* __restrict__ atte,
                                float* __restrict__ M) {
  int idx = threadIdx.x;
  if (idx >= 128) return;
  int kin = idx >> 3, h = idx & 7;
  float s = 0.f;
  for (int c = 0; c < 16; ++c) s += We[(h * 16 + c) * 16 + kin] * atte[h * 16 + c];
  M[kin * 8 + h] = s;
}

// ---------------- WMMA GEMM: out[m] = h16 @ wt[m]  (3 matrices via blockIdx.y) ----------------
// Outputs are NPAD rows -> no store guards needed (pad rows are zero and never read).
__global__ __launch_bounds__(256)
void k_gemm_qkv(const _Float16* __restrict__ h16, const _Float16* __restrict__ wt,
                float* __restrict__ q, float* __restrict__ kk, float* __restrict__ v) {
  __shared__ _Float16 sW[128 * 128];          // 32 KB of the 320 KB/WGP LDS
  const int tid = threadIdx.x;
  const int mat = blockIdx.y;
  {
    const uint32_t* src = (const uint32_t*)(wt + (size_t)mat * 16384);
    uint32_t* dst = (uint32_t*)sW;
    for (int i = tid; i < 128 * 128 / 2; i += 256) dst[i] = src[i];
  }
  __syncthreads();

  const int wave  = tid >> 5;
  const int lane  = tid & 31;
  const int mBase = blockIdx.x * 128 + wave * 16;
  const int aRow  = mBase + (lane & 15);
  const int g8    = (lane >> 4) * 8;
  float* out = (mat == 0) ? q : (mat == 1) ? kk : v;

  for (int jt = 0; jt < 8; ++jt) {
    const int jBase = jt * 16;
    v8f acc = {};
#pragma unroll
    for (int kc = 0; kc < 4; ++kc) {
      const int kBase = kc * 32;
      // A fragment: 16x32 f16, lane = M (mod 16), VGPR r holds K pair per ISA table
      ABFrag a;
      const uint32_t* arow = (const uint32_t*)(h16 + (size_t)aRow * 128 + kBase);
#pragma unroll
      for (int r = 0; r < 8; ++r) {
        int kpair = (r < 4) ? (2 * r + g8) : (2 * (r - 4) + 16 + g8);
        a.i[r] = (int)arow[kpair >> 1];
      }
      // B fragment: 32x16 f16, lane = N (mod 16), element s holds K per ISA table
      ABFrag b;
      const _Float16* bcol = sW + (size_t)kBase * 128 + jBase + (lane & 15);
#pragma unroll
      for (int s = 0; s < 16; ++s) b.h[s] = bcol[(size_t)((s < 8) ? (s + g8) : (s - 8 + 16 + g8)) * 128];

      acc = __builtin_amdgcn_wmma_f32_16x16x32_f16(false, a.h, false, b.h,
                                                   (short)0, acc, false, false);
    }
    // D: lane = N (mod 16), VGPR r -> M = r + 8*(lane/16); unguarded (padded output)
    const int n    = lane & 15;
    const int mOff = (lane >> 4) * 8;
    float* obase = out + (size_t)(mBase + mOff) * 128 + jBase + n;
#pragma unroll
    for (int r = 0; r < 8; ++r) obase[(size_t)r * 128] = acc[r];
  }
}

// ---------------- edge pipeline ----------------
__global__ void k_eterm(const float* __restrict__ ea, const float* __restrict__ M,
                        float* __restrict__ et, int E) {
  int e = blockIdx.x * blockDim.x + threadIdx.x;
  if (e >= E) return;
  float x[16];
#pragma unroll
  for (int c = 0; c < 16; ++c) x[c] = ea[(size_t)e * 16 + c];
#pragma unroll
  for (int h = 0; h < 8; ++h) {
    float s = 0.f;
#pragma unroll
    for (int c = 0; c < 16; ++c) s += x[c] * M[c * 8 + h];
    et[(size_t)e * 8 + h] = s;
  }
}

__global__ void k_alpha(const float* __restrict__ qs, const float* __restrict__ ks,
                        const float* __restrict__ et, const int* __restrict__ ei,
                        float* __restrict__ a, unsigned* __restrict__ amaxKey, int E) {
  int idx = blockIdx.x * blockDim.x + threadIdx.x;
  if (idx >= E * 8) return;
  int e = idx >> 3, h = idx & 7;
  int src = ei[e], dst = ei[E + e];
  const f32x4* qp = (const f32x4*)(qs + (size_t)dst * 128 + h * 16);
  const f32x4* kp = (const f32x4*)(ks + (size_t)src * 128 + h * 16);
  float s = 0.f;
#pragma unroll
  for (int c4 = 0; c4 < 4; ++c4) {
    f32x4 qv = qp[c4], kv = kp[c4];
    s += qv.x * kv.x + qv.y * kv.y + qv.z * kv.z + qv.w * kv.w;
  }
  s += et[idx];
  s = (s > 0.f) ? s : LEAKY * s;
  a[idx] = s;
  atomicMax(amaxKey + (size_t)dst * 8 + h, f2key(s));
}

__global__ void k_expsum(float* __restrict__ a, const unsigned* __restrict__ amaxKey,
                         float* __restrict__ asum, const int* __restrict__ ei, int E) {
  int idx = blockIdx.x * blockDim.x + threadIdx.x;
  if (idx >= E * 8) return;
  int e = idx >> 3, h = idx & 7;
  int dst = ei[E + e];
  float m = key2f(amaxKey[(size_t)dst * 8 + h]);
  float v = expf(a[idx] - m);
  a[idx] = v;
  atomicAdd(asum + (size_t)dst * 8 + h, v);
}

// aggregation with fused softmax normalization: thread per (edge, 4 channels)
__global__ void k_aggregate(const float* __restrict__ a, const float* __restrict__ asum,
                            const float* __restrict__ v, const int* __restrict__ ei,
                            float* __restrict__ agg, int E) {
  int idx = blockIdx.x * blockDim.x + threadIdx.x;
  if (idx >= E * 32) return;
  int e = idx >> 5, t = idx & 31;   // t: which float4 of the 128-channel row
  int h = t >> 2;                   // head = channel/16
  int src = ei[e], dst = ei[E + e];
  float w = a[(size_t)e * 8 + h] / (asum[(size_t)dst * 8 + h] + 1e-16f);
  f32x4 vv = *(const f32x4*)(v + (size_t)src * 128 + t * 4);
  float* o = agg + (size_t)dst * 128 + t * 4;
  atomicAdd(o + 0, w * vv.x);
  atomicAdd(o + 1, w * vv.y);
  atomicAdd(o + 2, w * vv.z);
  atomicAdd(o + 3, w * vv.w);
}

// ---------------- batchnorm over N x 128 ----------------
__global__ __launch_bounds__(128)
void k_bn_stats(const float* __restrict__ agg, float* __restrict__ sums, int nRows) {
  int j  = threadIdx.x;                 // column
  int r0 = blockIdx.x * 256;
  int r1 = min(r0 + 256, nRows);
  float s = 0.f, s2 = 0.f;
  for (int r = r0; r < r1; ++r) {
    float x = agg[(size_t)r * 128 + j];
    s += x; s2 += x * x;
  }
  atomicAdd(sums + j, s);
  atomicAdd(sums + 128 + j, s2);
}

__global__ void k_bn_norm(const float* __restrict__ agg, const float* __restrict__ sums,
                          const float* __restrict__ sc, const float* __restrict__ sh,
                          float* __restrict__ h32, _Float16* __restrict__ h16,
                          int nRows, int doRelu) {
  int idx = blockIdx.x * blockDim.x + threadIdx.x;
  if (idx >= nRows * 128) return;
  int j = idx & 127;
  float mu  = sums[j] / (float)nRows;
  float var = sums[128 + j] / (float)nRows - mu * mu;
  float y = (agg[idx] - mu) * rsqrtf(var + BNEPS) * sc[j] + sh[j];
  if (doRelu) y = fmaxf(y, 0.f);
  h32[idx] = y;
  h16[idx] = (_Float16)y;
}

// ---------------- pooling + MLP head ----------------
__global__ void k_pool(const float* __restrict__ h32, const int* __restrict__ batch,
                       unsigned* __restrict__ gkey, int nRows) {
  int idx = blockIdx.x * blockDim.x + threadIdx.x;
  if (idx >= nRows * 128) return;
  int n = idx >> 7, j = idx & 127;
  atomicMax(gkey + (size_t)batch[n] * 128 + j, f2key(h32[idx]));
}

__global__ void k_decode(const unsigned* __restrict__ gkey, float* __restrict__ gp, int n) {
  int i = blockIdx.x * blockDim.x + threadIdx.x;
  if (i < n) gp[i] = key2f(gkey[i]);
}

__global__ void k_linear(const float* __restrict__ X, const float* __restrict__ W,
                         const float* __restrict__ b, float* __restrict__ Y,
                         int G, int J, int K) {
  int idx = blockIdx.x * blockDim.x + threadIdx.x;
  if (idx >= G * J) return;
  int g = idx / J, j = idx % J;
  float s = b[j];
  for (int k = 0; k < K; ++k) s += X[(size_t)g * K + k] * W[(size_t)j * K + k];
  Y[idx] = s;
}

__global__ void k_bn_small(float* __restrict__ Z, const float* __restrict__ sc,
                           const float* __restrict__ sh, int rows, int cols, int doRelu) {
  int j = blockIdx.x;        // column
  int t = threadIdx.x;       // row (rows == 64)
  __shared__ float s1[64], s2[64];
  float x = Z[(size_t)t * cols + j];
  s1[t] = x; s2[t] = x * x;
  __syncthreads();
  for (int o = rows >> 1; o > 0; o >>= 1) {
    if (t < o) { s1[t] += s1[t + o]; s2[t] += s2[t + o]; }
    __syncthreads();
  }
  float mu  = s1[0] / (float)rows;
  float var = s2[0] / (float)rows - mu * mu;
  float y = (x - mu) * rsqrtf(var + BNEPS) * sc[j] + sh[j];
  if (doRelu) y = fmaxf(y, 0.f);
  Z[(size_t)t * cols + j] = y;
}

// ---------------- host orchestration ----------------
static inline int ceildiv(long long a, long long b) { return (int)((a + b - 1) / b); }

extern "C" void kernel_launch(void* const* d_in, const int* in_sizes, int n_in,
                              void* d_out, int out_size, void* d_ws, size_t ws_size,
                              hipStream_t stream) {
  (void)in_sizes; (void)n_in; (void)out_size; (void)ws_size;
  const float* x       = (const float*)d_in[0];
  const int*   ei      = (const int*)d_in[1];
  const float* ea      = (const float*)d_in[2];
  const int*   batch   = (const int*)d_in[3];
  const float* lin_qkv = (const float*)d_in[4];
  const float* lin_edge= (const float*)d_in[5];
  const float* att     = (const float*)d_in[6];
  const float* bn_sc   = (const float*)d_in[8];
  const float* bn_sh   = (const float*)d_in[9];
  const float* W1 = (const float*)d_in[10]; const float* b1 = (const float*)d_in[11];
  const float* s1 = (const float*)d_in[12]; const float* h1 = (const float*)d_in[13];
  const float* W2 = (const float*)d_in[14]; const float* b2 = (const float*)d_in[15];
  const float* s2 = (const float*)d_in[16]; const float* h2 = (const float*)d_in[17];
  const float* Wo = (const float*)d_in[18]; const float* bo = (const float*)d_in[19];
  float* out = (float*)d_out;

  // workspace carve-up
  char* ws = (char*)d_ws;
  size_t off = 0;
  auto carve = [&](size_t bytes) { void* p = ws + off; off = (off + bytes + 255) & ~(size_t)255; return p; };
  _Float16* h16    = (_Float16*)carve((size_t)NPAD * 128 * 2);
  float*    qs     = (float*)   carve((size_t)NPAD * 128 * 4);   // padded: unguarded WMMA stores
  float*    ks     = (float*)   carve((size_t)NPAD * 128 * 4);
  float*    vv     = (float*)   carve((size_t)NPAD * 128 * 4);
  float*    agg    = (float*)   carve((size_t)NNODES * 128 * 4);
  float*    h32    = (float*)   carve((size_t)NNODES * 128 * 4);
  float*    abuf   = (float*)   carve((size_t)NEDGES * 8 * 4);
  float*    eterm  = (float*)   carve((size_t)NEDGES * 8 * 4);
  unsigned* amaxK  = (unsigned*)carve((size_t)NNODES * 8 * 4);
  float*    asum   = (float*)   carve((size_t)NNODES * 8 * 4);
  _Float16* wt     = (_Float16*)carve((size_t)3 * 128 * 128 * 2);
  float*    Mmat   = (float*)   carve(16 * 8 * 4);
  float*    bnsum  = (float*)   carve(256 * 4);
  unsigned* gkey   = (unsigned*)carve((size_t)NGRAPH * 128 * 4);
  float*    gpool  = (float*)   carve((size_t)NGRAPH * 128 * 4);
  float*    z1     = (float*)   carve((size_t)NGRAPH * 64 * 4);
  float*    z2     = (float*)   carve((size_t)NGRAPH * 32 * 4);

  const int B = 256;
  // h16 = f16(x), padded rows zero
  k_fill_u32<<<ceildiv((size_t)NPAD * 64, B), B, 0, stream>>>((uint32_t*)h16, 0u, NPAD * 64);
  k_cvt_f16<<<ceildiv((size_t)NNODES * 128, B), B, 0, stream>>>(x, h16, NNODES * 128);

  for (int l = 0; l < 3; ++l) {
    const float* Wl   = lin_qkv + (size_t)l * 3 * 128 * 128;
    const float* attl = att + (size_t)l * 384;
    const float* Wel  = lin_edge + (size_t)l * 128 * 16;

    k_prep_qkv_w<<<ceildiv(3 * 16384, B), B, 0, stream>>>(Wl, attl, wt);
    k_prep_edge_mat<<<1, 128, 0, stream>>>(Wel, attl + 256, Mmat);

    dim3 gg(NPAD / 128, 3);
    k_gemm_qkv<<<gg, B, 0, stream>>>(h16, wt, qs, ks, vv);

    k_eterm<<<ceildiv(NEDGES, B), B, 0, stream>>>(ea, Mmat, eterm, NEDGES);

    k_fill_u32<<<ceildiv(NNODES * 8, B), B, 0, stream>>>(amaxK, NEGKEY, NNODES * 8);
    k_alpha<<<ceildiv((size_t)NEDGES * 8, B), B, 0, stream>>>(qs, ks, eterm, ei, abuf, amaxK, NEDGES);

    k_fill_u32<<<ceildiv(NNODES * 8, B), B, 0, stream>>>((uint32_t*)asum, 0u, NNODES * 8);
    k_expsum<<<ceildiv((size_t)NEDGES * 8, B), B, 0, stream>>>(abuf, amaxK, asum, ei, NEDGES);

    k_fill_u32<<<ceildiv((size_t)NNODES * 128, B), B, 0, stream>>>((uint32_t*)agg, 0u, NNODES * 128);
    k_aggregate<<<ceildiv((size_t)NEDGES * 32, B), B, 0, stream>>>(abuf, asum, vv, ei, agg, NEDGES);

    k_fill_u32<<<1, 256, 0, stream>>>((uint32_t*)bnsum, 0u, 256);
    k_bn_stats<<<ceildiv(NNODES, 256), 128, 0, stream>>>(agg, bnsum, NNODES);
    k_bn_norm<<<ceildiv((size_t)NNODES * 128, B), B, 0, stream>>>(
        agg, bnsum, bn_sc + l * 128, bn_sh + l * 128, h32, h16, NNODES, (l < 2) ? 1 : 0);
  }

  // global max pool -> [64,128]
  k_fill_u32<<<ceildiv(NGRAPH * 128, B), B, 0, stream>>>(gkey, NEGKEY, NGRAPH * 128);
  k_pool<<<ceildiv((size_t)NNODES * 128, B), B, 0, stream>>>(h32, batch, gkey, NNODES);
  k_decode<<<ceildiv(NGRAPH * 128, B), B, 0, stream>>>(gkey, gpool, NGRAPH * 128);

  // MLP head
  k_linear<<<ceildiv(NGRAPH * 64, B), B, 0, stream>>>(gpool, W1, b1, z1, NGRAPH, 64, 128);
  k_bn_small<<<64, 64, 0, stream>>>(z1, s1, h1, NGRAPH, 64, 1);
  k_linear<<<ceildiv(NGRAPH * 32, B), B, 0, stream>>>(z1, W2, b2, z2, NGRAPH, 32, 64);
  k_bn_small<<<32, 64, 0, stream>>>(z2, s2, h2, NGRAPH, 32, 1);
  k_linear<<<ceildiv(NGRAPH * 64, B), B, 0, stream>>>(z2, Wo, bo, out, NGRAPH, 64, 32);
}
